// NewsNet_52716428591486
// MI455X (gfx1250) — matrix operand order repaired and checked by reading, not compiled, not run on hardware
//
#include <hip/hip_runtime.h>

typedef __attribute__((ext_vector_type(8)))  _Float16 half8;
typedef __attribute__((ext_vector_type(16))) _Float16 v16h;
typedef __attribute__((ext_vector_type(8)))  float    v8f;
typedef __attribute__((address_space(3)))    _Float16 lds_half;

#define FDIM 256
#define BSTR 56                 // LDS column stride (halfs): 112B, 16 distinct banks, 16B aligned
#define RSQRT2 0.70710678118654752f

__device__ __forceinline__ v16h make_frag(half8 lo, half8 hi) {
  return __builtin_shufflevector(lo, hi, 0,1,2,3,4,5,6,7,8,9,10,11,12,13,14,15);
}

// ---------------------------------------------------------------------------
// Weight prep: f32 W[256,256] (rows=k, cols=n) -> f16 blocked-transposed
// Wblk[(k/32)*256 + n][k%32]: each 32-K slice is one contiguous 16KB block.
// ---------------------------------------------------------------------------
__global__ void k_prep_w(const float* __restrict__ W, _Float16* __restrict__ Wblk) {
  const int i = blockIdx.x * blockDim.x + threadIdx.x;   // 65536 elements
  if (i < FDIM * FDIM) {
    const int k = i >> 8;
    const int n = i & 255;
    Wblk[(size_t)((k >> 5) * FDIM + n) * 32 + (k & 31)] = (_Float16)W[i];
  }
}

// ---------------------------------------------------------------------------
// C[M,256] = A[M,256](f16) x W(f16 blocked, staged via async-DMA to LDS)
// optional rowAdd: C[row,:] += rowAdd[row/P,:]   (per-graph broadcast row)
// Block: 256 threads (8 waves). Each wave: 16-row strip, all 256 cols.
// Double-buffered LDS; GLOBAL_LOAD_ASYNC_TO_LDS_B128 staging (ASYNCcnt).
// ---------------------------------------------------------------------------
__global__ __launch_bounds__(256)
void k_gemm256(const _Float16* __restrict__ A, const _Float16* __restrict__ Wblk,
               float* __restrict__ C, const float* __restrict__ rowAdd,
               int M, int P)
{
  __shared__ _Float16 Blds[2][FDIM * BSTR];
  const int  tid    = threadIdx.x;
  const int  lane   = tid & 31;
  const int  wv     = tid >> 5;
  const int  mlo    = lane & 15;
  const int  hiHalf = (lane & 16) ? 1 : 0;
  const long rowBase = (long)blockIdx.x * 128 + wv * 16;
  long arow = rowBase + mlo;
  if (arow > M - 1) arow = M - 1;           // clamp (M always mult of 16 here)
  arow *= FDIM;

  v8f acc[16] = {};

  // stage K-slice s into LDS buffer b: 4 async 16B transfers per thread,
  // global side fully contiguous (lane-consecutive q => consecutive 16B)
  auto stage = [&](int b, int s) {
    const _Float16* gbase = Wblk + (size_t)s * (FDIM * 32);
    #pragma unroll
    for (int j = 0; j < 4; ++j) {
      const int q   = j * 256 + tid;        // 0..1023 transfer id
      const int col = q >> 2;
      const int seg = q & 3;
      const _Float16* gp = gbase + (size_t)q * 8;               // q*16 bytes
      lds_half* lp = (lds_half*)&Blds[b][col * BSTR + seg * 8]; // 16B aligned
      asm volatile("global_load_async_to_lds_b128 %0, %1, off"
                   :: "v"(lp), "v"(gp) : "memory");
    }
  };

  stage(0, 0);
  for (int s = 0; s < 8; ++s) {
    const int cur = s & 1;
    if (s + 1 < 8) {
      stage(cur ^ 1, s + 1);                 // overlap next slice with compute
      asm volatile("s_wait_asynccnt 0x4" ::: "memory");   // slice s retired (in-order)
    } else {
      asm volatile("s_wait_asynccnt 0x0" ::: "memory");
    }
    __syncthreads();                         // slice s visible block-wide

    // A fragment: ISA 16-bit A layout (16x32)
    const int k0 = s * 32;
    const int kA = k0 + (hiHalf ? 8 : 0);
    half8 alo = *(const half8*)(A + arow + kA);
    half8 ahi = *(const half8*)(A + arow + kA + 16);
    v16h  a   = make_frag(alo, ahi);

    #pragma unroll
    for (int ct = 0; ct < 16; ++ct) {
      // B fragment: lanes 0-15 col N=lane K 0..7,16..23; lanes 16-31 K 8..15,24..31
      const _Float16* bp = &Blds[cur][(ct * 16 + mlo) * BSTR + (hiHalf ? 8 : 0)];
      half8 blo = *(const half8*)bp;
      half8 bhi = *(const half8*)(bp + 16);
      v16h  b   = make_frag(blo, bhi);
      acc[ct] = __builtin_amdgcn_wmma_f32_16x16x32_f16(
          false, a, false, b, (short)0, acc[ct], false, false);
    }
    __syncthreads();                         // done reading buf[cur] before restage
  }

  #pragma unroll
  for (int ct = 0; ct < 16; ++ct) {
    const int col = ct * 16 + mlo;
    #pragma unroll
    for (int r = 0; r < 8; ++r) {
      const long row = rowBase + r + (hiHalf ? 8 : 0);
      if (row < M) {
        float v = acc[ct][r];
        if (rowAdd) v += rowAdd[(row / P) * FDIM + col];
        C[row * FDIM + col] = v;
      }
    }
  }
}

// ---------------------------------------------------------------------------
// small utility kernels
// ---------------------------------------------------------------------------
__global__ void k_zero_f32(float* p, long n) {
  long i = (long)blockIdx.x * blockDim.x + threadIdx.x;
  long s = (long)gridDim.x * blockDim.x;
  for (; i < n; i += s) p[i] = 0.0f;
}
__global__ void k_zero_i32(int* p, long n) {
  long i = (long)blockIdx.x * blockDim.x + threadIdx.x;
  long s = (long)gridDim.x * blockDim.x;
  for (; i < n; i += s) p[i] = 0;
}
__global__ void k_f32_to_f16(const float* __restrict__ in, _Float16* __restrict__ out, long n) {
  long i = (long)blockIdx.x * blockDim.x + threadIdx.x;
  long s = (long)gridDim.x * blockDim.x;
  for (; i < n; i += s) out[i] = (_Float16)in[i];
}
// rooth[g,n] = f16(relu(x[g*P, n]))
__global__ void k_rooth(const float* __restrict__ x, _Float16* __restrict__ rooth,
                        int G, int P) {
  long i = (long)blockIdx.x * blockDim.x + threadIdx.x;
  long n = (long)G * FDIM;
  long s = (long)gridDim.x * blockDim.x;
  for (; i < n; i += s) {
    long g = i >> 8, c = i & 255;
    float v = x[(g * (long)P) * FDIM + c];
    rooth[i] = (_Float16)(v > 0.0f ? v : 0.0f);
  }
}
// parent[dst]=src ; childcount[src]++   (tree: each child has exactly one edge)
__global__ void k_edges(const int* __restrict__ src, const int* __restrict__ dst,
                        int* __restrict__ parent, int* __restrict__ ccount, int E) {
  int e = blockIdx.x * blockDim.x + threadIdx.x;
  if (e < E) {
    parent[dst[e]] = src[e];
    atomicAdd(&ccount[src[e]], 1);
  }
}
__global__ void k_dinv_bu(const int* __restrict__ ccount, float* __restrict__ dinv, long n) {
  long i = (long)blockIdx.x * blockDim.x + threadIdx.x;
  long s = (long)gridDim.x * blockDim.x;
  for (; i < n; i += s) dinv[i] = rsqrtf((float)(ccount[i] + 1));
}

// ---------------------------------------------------------------------------
// top-down propagation: gather from parent (tree => single in-edge)
// mode 0: outHalf = f16(relu(h + b))      mode 1: atomicAdd pool[g,:] relu(h+b)
// ---------------------------------------------------------------------------
__global__ __launch_bounds__(256)
void k_prop_td(const float* __restrict__ xw, const int* __restrict__ parent,
               const float* __restrict__ bvec, _Float16* __restrict__ outHalf,
               float* __restrict__ pool, int mode, int P)
{
  const long v = blockIdx.x;
  const int  n = threadIdx.x;
  const bool root = (v % P) == 0;
  const float dv = root ? 1.0f : RSQRT2;
  float h = xw[v * FDIM + n] * dv * dv;
  if (!root) {
    const long p = parent[v];
    const float dp = (p % P == 0) ? 1.0f : RSQRT2;
    h += xw[p * FDIM + n] * dp * dv;
  }
  h += bvec[n];
  h = h > 0.0f ? h : 0.0f;
  if (mode == 0) outHalf[v * FDIM + n] = (_Float16)h;
  else           atomicAdd(&pool[(v / P) * FDIM + n], h);
}

// bottom-up: acc[v] = xw[v]*dinv[v]^2 (self loop term)
__global__ __launch_bounds__(256)
void k_init_bu(const float* __restrict__ xw, const float* __restrict__ dinv,
               float* __restrict__ acc)
{
  const long v = blockIdx.x;
  const int  n = threadIdx.x;
  const float d = dinv[v];
  acc[v * FDIM + n] = xw[v * FDIM + n] * d * d;
}
// bottom-up scatter: child v adds xw[v]*dinv[v]*dinv[parent] into acc[parent]
__global__ __launch_bounds__(256)
void k_scatter_bu(const float* __restrict__ xw, const int* __restrict__ parent,
                  const float* __restrict__ dinv, float* __restrict__ acc, int P)
{
  const long v = blockIdx.x;
  if ((v % P) == 0) return;                 // roots have no parent
  const int  n = threadIdx.x;
  const long p = parent[v];
  const float s = dinv[v] * dinv[p];
  atomicAdd(&acc[p * FDIM + n], xw[v * FDIM + n] * s);
}
__global__ __launch_bounds__(256)
void k_epi_bu(const float* __restrict__ acc, const float* __restrict__ bvec,
              _Float16* __restrict__ outHalf, float* __restrict__ pool,
              int mode, int P)
{
  const long v = blockIdx.x;
  const int  n = threadIdx.x;
  float h = acc[v * FDIM + n] + bvec[n];
  h = h > 0.0f ? h : 0.0f;
  if (mode == 0) outHalf[v * FDIM + n] = (_Float16)h;
  else           atomicAdd(&pool[(v / P) * FDIM + n], h);
}

// final: mean-pool already summed -> /P, linear [512,4], log_softmax
__global__ void k_final(const float* __restrict__ poolTd, const float* __restrict__ poolBu,
                        const float* __restrict__ Wl, const float* __restrict__ bl,
                        float* __restrict__ out, int G, int P)
{
  const int g = blockIdx.x * blockDim.x + threadIdx.x;
  if (g >= G) return;
  const float invP = 1.0f / (float)P;
  float l[4] = {bl[0], bl[1], bl[2], bl[3]};
  for (int j = 0; j < FDIM; ++j) {
    const float t = poolTd[g * FDIM + j] * invP;
    #pragma unroll
    for (int c = 0; c < 4; ++c) l[c] += t * Wl[j * 4 + c];
  }
  for (int j = 0; j < FDIM; ++j) {
    const float t = poolBu[g * FDIM + j] * invP;
    #pragma unroll
    for (int c = 0; c < 4; ++c) l[c] += t * Wl[(FDIM + j) * 4 + c];
  }
  float m = l[0];
  #pragma unroll
  for (int c = 1; c < 4; ++c) m = l[c] > m ? l[c] : m;
  float sum = 0.0f;
  #pragma unroll
  for (int c = 0; c < 4; ++c) sum += __expf(l[c] - m);
  const float lse = m + __logf(sum);
  #pragma unroll
  for (int c = 0; c < 4; ++c) out[g * 4 + c] = l[c] - lse;
}

// ---------------------------------------------------------------------------
extern "C" void kernel_launch(void* const* d_in, const int* in_sizes, int n_in,
                              void* d_out, int out_size, void* d_ws, size_t ws_size,
                              hipStream_t stream) {
  const float* x    = (const float*)d_in[0];
  const int*   ei   = (const int*)d_in[1];
  const float* W1td = (const float*)d_in[4];
  const float* b1td = (const float*)d_in[5];
  const float* W1bu = (const float*)d_in[6];
  const float* b1bu = (const float*)d_in[7];
  const float* W2td = (const float*)d_in[8];
  const float* b2td = (const float*)d_in[9];
  const float* W2bu = (const float*)d_in[10];
  const float* b2bu = (const float*)d_in[11];
  const float* Wl   = (const float*)d_in[12];
  const float* bl   = (const float*)d_in[13];
  float* out = (float*)d_out;

  const long N = (long)in_sizes[0] / FDIM;     // 131072
  const int  E = in_sizes[1] / 2;              // 130944
  const int  G = out_size / 4;                 // 128
  const int  P = (int)(N / G);                 // 1024

  // workspace carve-out (256B aligned)
  char* ws = (char*)d_ws;
  auto carve = [&](size_t bytes) -> char* {
    char* p = ws; ws += (bytes + 255) & ~(size_t)255; return p;
  };
  const size_t WBYTES = (size_t)FDIM * FDIM * 2;   // one blocked f16 weight
  _Float16* xh    = (_Float16*)carve((size_t)N * FDIM * 2);
  _Float16* a1    = (_Float16*)carve((size_t)N * FDIM * 2);
  float*    xw    = (float*)   carve((size_t)N * FDIM * 4);
  float*    acc   = (float*)   carve((size_t)N * FDIM * 4);
  _Float16* rooth = (_Float16*)carve((size_t)G * FDIM * 2);
  float*    Rtd   = (float*)   carve((size_t)G * FDIM * 4);
  float*    Rbu   = (float*)   carve((size_t)G * FDIM * 4);
  float*    poolTd= (float*)   carve((size_t)G * FDIM * 4);
  float*    poolBu= (float*)   carve((size_t)G * FDIM * 4);
  int*      parent= (int*)     carve((size_t)N * 4);
  int*      ccnt  = (int*)     carve((size_t)N * 4);
  float*    dinvBu= (float*)   carve((size_t)N * 4);
  _Float16* Wb1td = (_Float16*)carve(WBYTES);
  _Float16* Wb1bu = (_Float16*)carve(WBYTES);
  _Float16* Wb2tda= (_Float16*)carve(WBYTES);  // W2_td rows 0..255
  _Float16* Wb2tdb= (_Float16*)carve(WBYTES);  // W2_td rows 256..511 (root part)
  _Float16* Wb2bua= (_Float16*)carve(WBYTES);
  _Float16* Wb2bub= (_Float16*)carve(WBYTES);

  const int* srcE = ei;
  const int* dstE = ei + E;

  const long NE = N * FDIM;
  const int  T  = 256;
  const int  gElem = 2048;
  const int  wBlocks = (FDIM * FDIM + T - 1) / T;
  const int  gemmBlocksN = (int)((N + 127) / 128);
  const int  gemmBlocksG = (G + 127) / 128;

  // ---- graph structure ----
  k_zero_i32<<<gElem, T, 0, stream>>>(ccnt, N);
  k_edges<<<(E + T - 1) / T, T, 0, stream>>>(srcE, dstE, parent, ccnt, E);
  k_dinv_bu<<<gElem, T, 0, stream>>>(ccnt, dinvBu, N);

  // ---- precision staging ----
  k_f32_to_f16<<<gElem, T, 0, stream>>>(x, xh, NE);
  k_rooth<<<(G * FDIM + T - 1) / T, T, 0, stream>>>(x, rooth, G, P);
  k_prep_w<<<wBlocks, T, 0, stream>>>(W1td, Wb1td);
  k_prep_w<<<wBlocks, T, 0, stream>>>(W1bu, Wb1bu);
  k_prep_w<<<wBlocks, T, 0, stream>>>(W2td, Wb2tda);
  k_prep_w<<<wBlocks, T, 0, stream>>>(W2td + (size_t)FDIM * FDIM, Wb2tdb);
  k_prep_w<<<wBlocks, T, 0, stream>>>(W2bu, Wb2bua);
  k_prep_w<<<wBlocks, T, 0, stream>>>(W2bu + (size_t)FDIM * FDIM, Wb2bub);

  // ---- per-graph root GEMMs: R = relu(x[root]) @ W2[256:512,:] ----
  k_gemm256<<<gemmBlocksG, T, 0, stream>>>(rooth, Wb2tdb, Rtd, nullptr, G, P);
  k_gemm256<<<gemmBlocksG, T, 0, stream>>>(rooth, Wb2bub, Rbu, nullptr, G, P);

  // ================= top-down pipeline =================
  k_gemm256<<<gemmBlocksN, T, 0, stream>>>(xh, Wb1td, xw, nullptr, (int)N, P);
  k_prop_td<<<(int)N, T, 0, stream>>>(xw, parent, b1td, a1, nullptr, 0, P);
  k_gemm256<<<gemmBlocksN, T, 0, stream>>>(a1, Wb2tda, xw, Rtd, (int)N, P);
  k_zero_f32<<<(G * FDIM + T - 1) / T, T, 0, stream>>>(poolTd, (long)G * FDIM);
  k_prop_td<<<(int)N, T, 0, stream>>>(xw, parent, b2td, nullptr, poolTd, 1, P);

  // ================= bottom-up pipeline =================
  k_gemm256<<<gemmBlocksN, T, 0, stream>>>(xh, Wb1bu, xw, nullptr, (int)N, P);
  k_init_bu<<<(int)N, T, 0, stream>>>(xw, dinvBu, acc);
  k_scatter_bu<<<(int)N, T, 0, stream>>>(xw, parent, dinvBu, acc, P);
  k_epi_bu<<<(int)N, T, 0, stream>>>(acc, b1bu, a1, nullptr, 0, P);
  k_gemm256<<<gemmBlocksN, T, 0, stream>>>(a1, Wb2bua, xw, Rbu, (int)N, P);
  k_zero_f32<<<(G * FDIM + T - 1) / T, T, 0, stream>>>(poolBu, (long)G * FDIM);
  k_init_bu<<<(int)N, T, 0, stream>>>(xw, dinvBu, acc);
  k_scatter_bu<<<(int)N, T, 0, stream>>>(xw, parent, dinvBu, acc, P);
  k_epi_bu<<<(int)N, T, 0, stream>>>(acc, b2bu, nullptr, poolBu, 1, P);

  // ---- classifier head ----
  k_final<<<(G + T - 1) / T, T, 0, stream>>>(poolTd, poolBu, Wl, bl, out, G, P);
}
